// ENFlow_44779329028665
// MI455X (gfx1250) — matrix operand order repaired and checked by reading, not compiled, not run on hardware
//
#include <hip/hip_runtime.h>

// ---------------------------------------------------------------------------
// ENFlow (E(n)-equivariant GNN flow) on gfx1250.
// Edge/node MLPs mapped to v_wmma_f32_16x16x32_f16 (16-row tiles); the 64->1
// head GEMVs (c2W, q2W) run as WMMA column tiles. Per-workgroup weight tiles
// staged global->LDS with global_load_async_to_lds_b128 (ASYNCcnt). Segment
// sums via global_atomic_add_f32; elementwise updates fused; SiLU via rcp.
// ---------------------------------------------------------------------------

typedef __attribute__((ext_vector_type(16))) _Float16 v16h;
typedef __attribute__((ext_vector_type(8)))  float    v8f;

#define DT_F  0.01f
#define BOX_F 20.0f

// Packed f16 weight-fragment offsets (in halfs) within one layer's block.
// Each fragment = 32 lanes * 16 halfs = 512 halfs; lane L reads its 16 halfs
// contiguously at frag_base + L*16 (two b128 loads).
// Edge group [0, 11264) ; node group [11264, 25600)
#define OFF_E1 0        //  e1: K 17->32 (1 chunk) x 4 N-tiles       =  4 frags
#define OFF_E2 2048     //  e2: K 64    (2)        x 4               =  8
#define OFF_C1 6144     //  c1: K 64    (2)        x 4               =  8
#define OFF_C2 10240    //  c2: K 64    (2)        x 1 (N 1->16)     =  2
#define OFF_N1 11264    //  n1: K 72->96(3)        x 4               = 12
#define OFF_N2 17408    //  n2: K 64    (2)        x 1 (N 8->16)     =  2
#define OFF_Q1 18432    //  q1: K 72->96(3)        x 4               = 12
#define OFF_Q2 24576    //  q2: K 64    (2)        x 1 (N 1->16)     =  2
#define LAYER_W_HALFS 25600
#define FRAGS_PER_LAYER 50

#define EDGE_W_HALFS 11264   // e1+e2+c1+c2
#define NODE_W_HALFS 14336   // n1+n2+q1+q2
// node-group offsets relative to staged node region
#define NR_N1 0
#define NR_N2 6144
#define NR_Q1 7168
#define NR_Q2 13312

#define WAVES_PB 8           // waves per block (256 threads)
#define EDGE_TPW 4           // edge tiles per wave
#define NODE_TPW 1           // node tiles per wave

// 16-bit A-matrix 16x32 per-lane layout (contiguous k, matching packed B): element j (0..15) of the
// v16h held by lane L (m=L%16, half=L/16) is A[m][ half*16 + j ].
__device__ __forceinline__ int akidx(int j, int half) {
  return (half << 4) + j;
}

__device__ __forceinline__ float silu(float x) {
  return x * __builtin_amdgcn_rcpf(1.0f + __expf(-x));
}

// Wave-local LDS ordering (waves in a block run independent tiles; no need
// for the block-wide barrier inside the pipeline).
__device__ __forceinline__ void wave_lds_sync() {
  __builtin_amdgcn_wave_barrier();
  asm volatile("s_wait_dscnt 0x0" ::: "memory");
  __builtin_amdgcn_wave_barrier();
}

// Bulk global->LDS copy on the CDNA5 async data-mover path (ASYNCcnt).
__device__ __forceinline__ void stage_async(const void* gsrc, void* lds,
                                            int bytes, int tid, int nthreads) {
  for (int o = tid * 16; o < bytes; o += nthreads * 16) {
    unsigned ldsa = (unsigned)(uintptr_t)((char*)lds + o);
    asm volatile("global_load_async_to_lds_b128 %0, %1, %2"
                 :: "v"(ldsa), "v"(o), "s"(gsrc) : "memory");
  }
  asm volatile("s_wait_asynccnt 0x0" ::: "memory");
}

struct LayerW {
  const float *e1W, *e1b, *e2W, *e2b, *c1W, *c1b, *c2W;
  const float *n1W, *n1b, *n2W, *n2b, *q1W, *q1b, *q2W, *q2b;
};

// ---------------------------------------------------------------------------
// Pack one layer's f32 weights into f16 B-fragments (per-lane VGPR order).
// B 32x16 f16 order used here (contiguous k, as akidx): lane L (n=L%16, kgrp=L/16) holds K = kgrp*16 + j.
// ---------------------------------------------------------------------------
__global__ __launch_bounds__(32) void pack_weights(LayerW lw, _Float16* dst) {
  int f = blockIdx.x, lane = threadIdx.x;
  const float* W; int Krows, Ncols, kc, t, off;
  if (f < 4)        { W = lw.e1W; Krows = 17; Ncols = 64; kc = 0;            t = f;            off = OFF_E1 + f * 512; }
  else if (f < 12)  { int r = f - 4;  W = lw.e2W; Krows = 64; Ncols = 64; kc = r >> 2; t = r & 3; off = OFF_E2 + r * 512; }
  else if (f < 20)  { int r = f - 12; W = lw.c1W; Krows = 64; Ncols = 64; kc = r >> 2; t = r & 3; off = OFF_C1 + r * 512; }
  else if (f < 22)  { int r = f - 20; W = lw.c2W; Krows = 64; Ncols = 1;  kc = r;      t = 0;     off = OFF_C2 + r * 512; }
  else if (f < 34)  { int r = f - 22; W = lw.n1W; Krows = 72; Ncols = 64; kc = r >> 2; t = r & 3; off = OFF_N1 + r * 512; }
  else if (f < 36)  { int r = f - 34; W = lw.n2W; Krows = 64; Ncols = 8;  kc = r;      t = 0;     off = OFF_N2 + r * 512; }
  else if (f < 48)  { int r = f - 36; W = lw.q1W; Krows = 72; Ncols = 64; kc = r >> 2; t = r & 3; off = OFF_Q1 + r * 512; }
  else              { int r = f - 48; W = lw.q2W; Krows = 64; Ncols = 1;  kc = r;      t = 0;     off = OFF_Q2 + r * 512; }
  int n = t * 16 + (lane & 15);
  int kbase = kc * 32 + ((lane >> 4) << 4);
#pragma unroll
  for (int j = 0; j < 16; ++j) {
    int K = kbase + j;
    float v = (K < Krows && n < Ncols) ? W[K * Ncols + n] : 0.0f;
    dst[off + lane * 16 + j] = (_Float16)v;
  }
}

// ---------------------------------------------------------------------------
// Edge kernel: 8 waves/block, each wave owns a 16-edge tile pipeline.
//   eh=[h_s|h_d|r2|0...] -> e1(silu) -> e2(silu)=m -> c1(silu) -> c2 (wmma col)
//   scatter: F[src] += diff*c ; agg[src] += m      (global f32 atomics)
// ---------------------------------------------------------------------------
__global__ __launch_bounds__(256) void edge_kernel(
    const float* __restrict__ hs, const float* __restrict__ poss,
    const int* __restrict__ esrc, const int* __restrict__ edst,
    const _Float16* __restrict__ wl,
    const float* __restrict__ e1b, const float* __restrict__ e2b,
    const float* __restrict__ c1b,
    float* __restrict__ Facc, float* __restrict__ agg, int Ee) {
  __shared__ _Float16 wlds[EDGE_W_HALFS];
  __shared__ _Float16 eh [WAVES_PB][16 * 32];
  __shared__ _Float16 bm [WAVES_PB][16 * 64];
  __shared__ _Float16 bm2[WAVES_PB][16 * 64];
  __shared__ _Float16 bch[WAVES_PB][16 * 64];
  __shared__ float    cv  [WAVES_PB][16];
  __shared__ int      ssrc[WAVES_PB][16];

  const int tid = threadIdx.x;
  const int wave = tid >> 5, lane = tid & 31;
  const int m = lane & 15, half = lane >> 4;

  stage_async(wl, wlds, EDGE_W_HALFS * 2, tid, 256);

  // Hoist loop-invariant per-lane biases into registers (the asm "memory"
  // clobbers in the pipeline otherwise force reloads every tile).
  float be1[4], be2[4], bc1[4];
#pragma unroll
  for (int t = 0; t < 4; ++t) {
    be1[t] = e1b[t * 16 + m];
    be2[t] = e2b[t * 16 + m];
    bc1[t] = c1b[t * 16 + m];
  }
  __syncthreads();

  _Float16* EH = eh[wave];
  _Float16* BM = bm[wave];
  _Float16* BM2 = bm2[wave];
  _Float16* BCH = bch[wave];

  const int tiles = (Ee + 15) >> 4;
  for (int tile = blockIdx.x * WAVES_PB + wave; tile < tiles;
       tile += gridDim.x * WAVES_PB) {
    float d0 = 0.f, d1 = 0.f, d2 = 0.f;
    int sidx = -1;

    if (half == 0) {
      int e = tile * 16 + m;
      if (e < Ee) {
        int s = esrc[e], d = edst[e];
        sidx = s;
        d0 = poss[s * 3 + 0] - poss[d * 3 + 0];
        d1 = poss[s * 3 + 1] - poss[d * 3 + 1];
        d2 = poss[s * 3 + 2] - poss[d * 3 + 2];
        float r2 = d0 * d0 + d1 * d1 + d2 * d2;
#pragma unroll
        for (int k = 0; k < 8; ++k) EH[m * 32 + k]     = (_Float16)hs[s * 8 + k];
#pragma unroll
        for (int k = 0; k < 8; ++k) EH[m * 32 + 8 + k] = (_Float16)hs[d * 8 + k];
        EH[m * 32 + 16] = (_Float16)r2;
#pragma unroll
        for (int k = 17; k < 32; ++k) EH[m * 32 + k] = (_Float16)0.f;
      } else {
#pragma unroll
        for (int k = 0; k < 32; ++k) EH[m * 32 + k] = (_Float16)0.f;
      }
      ssrc[wave][m] = sidx;
    }
    wave_lds_sync();

    // ---- e1: [16,32] x [32,16] x4 tiles -> BM (silu) ----
    v16h a1;
#pragma unroll
    for (int j = 0; j < 16; ++j) a1[j] = EH[m * 32 + akidx(j, half)];
#pragma unroll
    for (int t = 0; t < 4; ++t) {
      v16h b = *(const v16h*)(wlds + OFF_E1 + t * 512 + lane * 16);
      v8f acc = {};
      acc = __builtin_amdgcn_wmma_f32_16x16x32_f16(false, a1, false, b, (short)0, acc, false, false);
#pragma unroll
      for (int r = 0; r < 8; ++r)
        BM[(r + 8 * half) * 64 + t * 16 + m] = (_Float16)silu(acc[r] + be1[t]);
    }
    wave_lds_sync();

    // ---- e2: [16,64] x [64,16] x4 -> BM2 (silu) = messages m ----
    v16h a20, a21;
#pragma unroll
    for (int j = 0; j < 16; ++j) {
      a20[j] = BM[m * 64 +      akidx(j, half)];
      a21[j] = BM[m * 64 + 32 + akidx(j, half)];
    }
#pragma unroll
    for (int t = 0; t < 4; ++t) {
      v16h b0 = *(const v16h*)(wlds + OFF_E2 + (t)     * 512 + lane * 16);
      v16h b1 = *(const v16h*)(wlds + OFF_E2 + (4 + t) * 512 + lane * 16);
      v8f acc = {};
      acc = __builtin_amdgcn_wmma_f32_16x16x32_f16(false, a20, false, b0, (short)0, acc, false, false);
      acc = __builtin_amdgcn_wmma_f32_16x16x32_f16(false, a21, false, b1, (short)0, acc, false, false);
#pragma unroll
      for (int r = 0; r < 8; ++r)
        BM2[(r + 8 * half) * 64 + t * 16 + m] = (_Float16)silu(acc[r] + be2[t]);
    }
    wave_lds_sync();

    // ---- c1: BM2 -> BCH (silu) ----
    v16h a30, a31;
#pragma unroll
    for (int j = 0; j < 16; ++j) {
      a30[j] = BM2[m * 64 +      akidx(j, half)];
      a31[j] = BM2[m * 64 + 32 + akidx(j, half)];
    }
#pragma unroll
    for (int t = 0; t < 4; ++t) {
      v16h b0 = *(const v16h*)(wlds + OFF_C1 + (t)     * 512 + lane * 16);
      v16h b1 = *(const v16h*)(wlds + OFF_C1 + (4 + t) * 512 + lane * 16);
      v8f acc = {};
      acc = __builtin_amdgcn_wmma_f32_16x16x32_f16(false, a30, false, b0, (short)0, acc, false, false);
      acc = __builtin_amdgcn_wmma_f32_16x16x32_f16(false, a31, false, b1, (short)0, acc, false, false);
#pragma unroll
      for (int r = 0; r < 8; ++r)
        BCH[(r + 8 * half) * 64 + t * 16 + m] = (_Float16)silu(acc[r] + bc1[t]);
    }
    wave_lds_sync();

    // ---- c2 as WMMA column GEMV: BCH[16,64] x c2W[64,1->16] ----
    {
      v16h a40, a41;
#pragma unroll
      for (int j = 0; j < 16; ++j) {
        a40[j] = BCH[m * 64 +      akidx(j, half)];
        a41[j] = BCH[m * 64 + 32 + akidx(j, half)];
      }
      v16h b0 = *(const v16h*)(wlds + OFF_C2 + 0 * 512 + lane * 16);
      v16h b1 = *(const v16h*)(wlds + OFF_C2 + 1 * 512 + lane * 16);
      v8f accC = {};
      accC = __builtin_amdgcn_wmma_f32_16x16x32_f16(false, a40, false, b0, (short)0, accC, false, false);
      accC = __builtin_amdgcn_wmma_f32_16x16x32_f16(false, a41, false, b1, (short)0, accC, false, false);
      if (m == 0) {  // column 0 lanes (0 and 16) hold the per-row dots
#pragma unroll
        for (int r = 0; r < 8; ++r) cv[wave][r + 8 * half] = accC[r];
      }
    }
    wave_lds_sync();

    // ---- F scatter (edge-owner lanes) ----
    if (half == 0 && sidx >= 0) {
      float c = cv[wave][m];
      atomicAdd(&Facc[sidx * 3 + 0], d0 * c);
      atomicAdd(&Facc[sidx * 3 + 1], d1 * c);
      atomicAdd(&Facc[sidx * 3 + 2], d2 * c);
    }
    // ---- agg scatter: 16x64 values over 32 lanes ----
#pragma unroll
    for (int r = 0; r < 32; ++r) {
      int idx = r * 32 + lane;
      int row = idx >> 6, f = idx & 63;
      int s = ssrc[wave][row];
      if (s >= 0) atomicAdd(&agg[s * 64 + f], (float)BM2[row * 64 + f]);
    }
    wave_lds_sync();
  }
}

// ---------------------------------------------------------------------------
// Node kernel: 8 waves/block, each wave owns a 16-node tile pipeline.
//   nin=[h|agg|0...] -> n1(silu) -> n2 (G)        ; g += G*DT ; h += g*DT
//                    -> q1(silu) -> q2 (wmma col) ; vel=exp(Q)*vel+F*DT ; pos
// ---------------------------------------------------------------------------
__global__ __launch_bounds__(256) void node_kernel(
    float* __restrict__ hs, float* __restrict__ gs,
    float* __restrict__ poss, float* __restrict__ vels,
    const float* __restrict__ agg, const float* __restrict__ Facc,
    const _Float16* __restrict__ wl,
    const float* __restrict__ n1b, const float* __restrict__ n2b,
    const float* __restrict__ q1b, const float* __restrict__ q2b,
    float* __restrict__ ldj, int Nn) {
  __shared__ _Float16 wlds[NODE_W_HALFS];
  __shared__ _Float16 ni[WAVES_PB][16 * 96];
  __shared__ _Float16 bh[WAVES_PB][16 * 64];
  __shared__ float    qv[WAVES_PB][16];

  const int tid = threadIdx.x;
  const int wave = tid >> 5, lane = tid & 31;
  const int m = lane & 15, half = lane >> 4;

  stage_async(wl + OFF_N1, wlds, NODE_W_HALFS * 2, tid, 256);

  // Hoist loop-invariant per-lane biases into registers.
  float bn1[4], bq1[4];
#pragma unroll
  for (int t = 0; t < 4; ++t) {
    bn1[t] = n1b[t * 16 + m];
    bq1[t] = q1b[t * 16 + m];
  }
  const float bn2 = n2b[m & 7];
  const float bq2 = q2b[0];
  __syncthreads();

  _Float16* NI = ni[wave];
  _Float16* BH = bh[wave];

  const int tiles = (Nn + 15) >> 4;
  for (int tile = blockIdx.x * WAVES_PB + wave; tile < tiles;
       tile += gridDim.x * WAVES_PB) {
    const int base = tile * 16;
    const int node = base + m;
    const bool valid = node < Nn;

    if (half == 0) {
      if (valid) {
#pragma unroll
        for (int k = 0; k < 8; ++k)  NI[m * 96 + k]     = (_Float16)hs[node * 8 + k];
#pragma unroll
        for (int k = 0; k < 64; ++k) NI[m * 96 + 8 + k] = (_Float16)agg[node * 64 + k];
#pragma unroll
        for (int k = 72; k < 96; ++k) NI[m * 96 + k] = (_Float16)0.f;
      } else {
#pragma unroll
        for (int k = 0; k < 96; ++k) NI[m * 96 + k] = (_Float16)0.f;
      }
    }
    wave_lds_sync();

    v16h an0, an1, an2;
#pragma unroll
    for (int j = 0; j < 16; ++j) {
      an0[j] = NI[m * 96 +      akidx(j, half)];
      an1[j] = NI[m * 96 + 32 + akidx(j, half)];
      an2[j] = NI[m * 96 + 64 + akidx(j, half)];
    }

    // ---- n1 -> BH (silu) ----
#pragma unroll
    for (int t = 0; t < 4; ++t) {
      v16h b0 = *(const v16h*)(wlds + NR_N1 + (t)     * 512 + lane * 16);
      v16h b1 = *(const v16h*)(wlds + NR_N1 + (4 + t) * 512 + lane * 16);
      v16h b2 = *(const v16h*)(wlds + NR_N1 + (8 + t) * 512 + lane * 16);
      v8f acc = {};
      acc = __builtin_amdgcn_wmma_f32_16x16x32_f16(false, an0, false, b0, (short)0, acc, false, false);
      acc = __builtin_amdgcn_wmma_f32_16x16x32_f16(false, an1, false, b1, (short)0, acc, false, false);
      acc = __builtin_amdgcn_wmma_f32_16x16x32_f16(false, an2, false, b2, (short)0, acc, false, false);
#pragma unroll
      for (int r = 0; r < 8; ++r)
        BH[(r + 8 * half) * 64 + t * 16 + m] = (_Float16)silu(acc[r] + bn1[t]);
    }
    wave_lds_sync();

    // ---- n2 (64 -> 8, N padded to 16): G, then g/h updates ----
    v16h ah0, ah1;
#pragma unroll
    for (int j = 0; j < 16; ++j) {
      ah0[j] = BH[m * 64 +      akidx(j, half)];
      ah1[j] = BH[m * 64 + 32 + akidx(j, half)];
    }
    {
      v16h b0 = *(const v16h*)(wlds + NR_N2 + 0 * 512 + lane * 16);
      v16h b1 = *(const v16h*)(wlds + NR_N2 + 1 * 512 + lane * 16);
      v8f accG = {};
      accG = __builtin_amdgcn_wmma_f32_16x16x32_f16(false, ah0, false, b0, (short)0, accG, false, false);
      accG = __builtin_amdgcn_wmma_f32_16x16x32_f16(false, ah1, false, b1, (short)0, accG, false, false);
      if (m < 8) {
#pragma unroll
        for (int r = 0; r < 8; ++r) {
          int i = base + r + 8 * half;
          if (i < Nn) {
            float gn = gs[i * 8 + m] + DT_F * (accG[r] + bn2);
            gs[i * 8 + m] = gn;                 // g = g + G*dt
            hs[i * 8 + m] += DT_F * gn;         // h = h + g_new*dt
          }
        }
      }
    }
    wave_lds_sync();

    // ---- q1 -> BH (silu, reuse buffer) ----
#pragma unroll
    for (int t = 0; t < 4; ++t) {
      v16h b0 = *(const v16h*)(wlds + NR_Q1 + (t)     * 512 + lane * 16);
      v16h b1 = *(const v16h*)(wlds + NR_Q1 + (4 + t) * 512 + lane * 16);
      v16h b2 = *(const v16h*)(wlds + NR_Q1 + (8 + t) * 512 + lane * 16);
      v8f acc = {};
      acc = __builtin_amdgcn_wmma_f32_16x16x32_f16(false, an0, false, b0, (short)0, acc, false, false);
      acc = __builtin_amdgcn_wmma_f32_16x16x32_f16(false, an1, false, b1, (short)0, acc, false, false);
      acc = __builtin_amdgcn_wmma_f32_16x16x32_f16(false, an2, false, b2, (short)0, acc, false, false);
#pragma unroll
      for (int r = 0; r < 8; ++r)
        BH[(r + 8 * half) * 64 + t * 16 + m] = (_Float16)silu(acc[r] + bq1[t]);
    }
    wave_lds_sync();

    // ---- q2 as WMMA column GEMV ----
    {
      v16h aq0, aq1;
#pragma unroll
      for (int j = 0; j < 16; ++j) {
        aq0[j] = BH[m * 64 +      akidx(j, half)];
        aq1[j] = BH[m * 64 + 32 + akidx(j, half)];
      }
      v16h b0 = *(const v16h*)(wlds + NR_Q2 + 0 * 512 + lane * 16);
      v16h b1 = *(const v16h*)(wlds + NR_Q2 + 1 * 512 + lane * 16);
      v8f accQ = {};
      accQ = __builtin_amdgcn_wmma_f32_16x16x32_f16(false, aq0, false, b0, (short)0, accQ, false, false);
      accQ = __builtin_amdgcn_wmma_f32_16x16x32_f16(false, aq1, false, b1, (short)0, accQ, false, false);
      if (m == 0) {
#pragma unroll
        for (int r = 0; r < 8; ++r) qv[wave][r + 8 * half] = accQ[r];
      }
    }
    wave_lds_sync();

    // ---- symplectic updates (node-owner lanes) ----
    if (half == 0 && valid) {
      float Q = qv[wave][m] + bq2;
      float eQ = expf(Q);
#pragma unroll
      for (int k = 0; k < 3; ++k) {
        int ix = node * 3 + k;
        float v = eQ * vels[ix] + Facc[ix] * DT_F;   // vel = exp(Q)*vel + F*dt
        vels[ix] = v;
        float p = poss[ix] + v * DT_F;               // pos += vel*dt
        p = p - BOX_F * floorf(p / BOX_F);           // pbc wrap
        poss[ix] = p;
      }
      atomicAdd(ldj, Q);                             // ldj accumulates sum(Q)
    }
    wave_lds_sync();
  }
}

// ---------------------------------------------------------------------------
__global__ void finalize_kernel(const float* __restrict__ hs, const float* __restrict__ gs,
                                const float* __restrict__ ps, const float* __restrict__ vs,
                                const float* __restrict__ ldj, float* __restrict__ out, int Nn) {
  int h8 = Nn * 8, n3 = Nn * 3;
  int total = 2 * h8 + 2 * n3 + 1;
  for (int i = blockIdx.x * blockDim.x + threadIdx.x; i < total; i += gridDim.x * blockDim.x) {
    float v;
    if (i < h8)                 v = hs[i];
    else if (i < 2 * h8)        v = gs[i - h8];
    else if (i < 2 * h8 + n3)   v = ps[i - 2 * h8];
    else if (i < total - 1)     v = vs[i - 2 * h8 - n3];
    else                        v = *ldj;
    out[i] = v;
  }
}

// ---------------------------------------------------------------------------
extern "C" void kernel_launch(void* const* d_in, const int* in_sizes, int n_in,
                              void* d_out, int out_size, void* d_ws, size_t ws_size,
                              hipStream_t stream) {
  (void)out_size; (void)ws_size;
  const int N_ITER = 3;
  // Input order (setup_inputs insertion order, params leaves per layer in
  // definition order): h,g,pos,vel, 15x3 param leaves, edge_src, edge_dst.
  const float* in_h   = (const float*)d_in[0];
  const float* in_g   = (const float*)d_in[1];
  const float* in_pos = (const float*)d_in[2];
  const float* in_vel = (const float*)d_in[3];
  const int Nn = in_sizes[0] / 8;
  const int isrc = (n_in >= 2) ? n_in - 2 : 0;
  const int* e_src = (const int*)d_in[isrc];
  const int* e_dst = (const int*)d_in[isrc + 1];
  const int Ee = in_sizes[isrc];

  // Workspace layout (floats)
  float* fws = (float*)d_ws;
  size_t off = 0;
  float* w_h   = fws + off; off += (size_t)Nn * 8;
  float* w_g   = fws + off; off += (size_t)Nn * 8;
  float* w_pos = fws + off; off += (size_t)Nn * 3;
  float* w_vel = fws + off; off += (size_t)Nn * 3;
  float* w_F   = fws + off; off += (size_t)Nn * 3;   // F then agg: contiguous
  float* w_agg = fws + off; off += (size_t)Nn * 64;
  float* w_ldj = fws + off; off += 1;
  off = (off + 15) & ~(size_t)15;                     // 64B-align f16 region
  _Float16* w_wh = (_Float16*)(fws + off);

  // State in, ldj = 0
  hipMemcpyAsync(w_h,   in_h,   (size_t)Nn * 8 * sizeof(float), hipMemcpyDeviceToDevice, stream);
  hipMemcpyAsync(w_g,   in_g,   (size_t)Nn * 8 * sizeof(float), hipMemcpyDeviceToDevice, stream);
  hipMemcpyAsync(w_pos, in_pos, (size_t)Nn * 3 * sizeof(float), hipMemcpyDeviceToDevice, stream);
  hipMemcpyAsync(w_vel, in_vel, (size_t)Nn * 3 * sizeof(float), hipMemcpyDeviceToDevice, stream);
  hipMemsetAsync(w_ldj, 0, sizeof(float), stream);

  // Pack weights (f32 -> f16 B-fragments) once per launch
  LayerW LW[3];
  for (int l = 0; l < N_ITER; ++l) {
    const int b = 4 + 15 * l;
    LW[l].e1W = (const float*)d_in[b + 0];  LW[l].e1b = (const float*)d_in[b + 1];
    LW[l].e2W = (const float*)d_in[b + 2];  LW[l].e2b = (const float*)d_in[b + 3];
    LW[l].c1W = (const float*)d_in[b + 4];  LW[l].c1b = (const float*)d_in[b + 5];
    LW[l].c2W = (const float*)d_in[b + 6];
    LW[l].n1W = (const float*)d_in[b + 7];  LW[l].n1b = (const float*)d_in[b + 8];
    LW[l].n2W = (const float*)d_in[b + 9];  LW[l].n2b = (const float*)d_in[b + 10];
    LW[l].q1W = (const float*)d_in[b + 11]; LW[l].q1b = (const float*)d_in[b + 12];
    LW[l].q2W = (const float*)d_in[b + 13]; LW[l].q2b = (const float*)d_in[b + 14];
    pack_weights<<<FRAGS_PER_LAYER, 32, 0, stream>>>(LW[l], w_wh + (size_t)l * LAYER_W_HALFS);
  }

  const int eTiles = (Ee + 15) / 16;
  const int nTiles = (Nn + 15) / 16;
  const int eBlocks = (eTiles + WAVES_PB * EDGE_TPW - 1) / (WAVES_PB * EDGE_TPW);
  const int nBlocks = (nTiles + WAVES_PB * NODE_TPW - 1) / (WAVES_PB * NODE_TPW);
  for (int l = 0; l < N_ITER; ++l) {
    const _Float16* wl = w_wh + (size_t)l * LAYER_W_HALFS;
    hipMemsetAsync(w_F, 0, (size_t)Nn * 67 * sizeof(float), stream);  // F + agg
    edge_kernel<<<eBlocks, 256, 0, stream>>>(
        w_h, w_pos, e_src, e_dst, wl,
        LW[l].e1b, LW[l].e2b, LW[l].c1b,
        w_F, w_agg, Ee);
    node_kernel<<<nBlocks, 256, 0, stream>>>(
        w_h, w_g, w_pos, w_vel, w_agg, w_F, wl,
        LW[l].n1b, LW[l].n2b, LW[l].q1b, LW[l].q2b,
        w_ldj, Nn);
  }

  const int total = 2 * Nn * 8 + 2 * Nn * 3 + 1;
  finalize_kernel<<<(total + 255) / 256, 256, 0, stream>>>(
      w_h, w_g, w_pos, w_vel, w_ldj, (float*)d_out, Nn);
}
